// Separate_68573447847976
// MI455X (gfx1250) — matrix operand order repaired and checked by reading, not compiled, not run on hardware
//
#include <hip/hip_runtime.h>
#include <hip/hip_bf16.h>

// ---------------------------------------------------------------------------
// Types for WMMA fragments (gfx1250, wave32)
// ---------------------------------------------------------------------------
typedef __attribute__((ext_vector_type(16))) _Float16 v16h;
typedef __attribute__((ext_vector_type(8)))  _Float16 v8h;
typedef __attribute__((ext_vector_type(8)))  float    v8f;
typedef __attribute__((ext_vector_type(4)))  int      v4i;

union V16 { v16h v; v8h h[2]; };

__device__ __forceinline__ v16h ld16(const _Float16* p) {
  V16 t;
  t.h[0] = *(const v8h*)p;
  t.h[1] = *(const v8h*)(p + 8);
  return t.v;
}

__device__ __forceinline__ v8f wmma_f32(v16h a, v16h b, v8f c) {
  // D = A(16x32 f16) * B(32x16 f16) + C(16x16 f32)
  return __builtin_amdgcn_wmma_f32_16x16x32_f16(
      /*neg_a=*/false, a, /*neg_b=*/false, b,
      /*c_mod=*/(short)0, c, /*reuse_a=*/false, /*reuse_b=*/false);
}

// ---------------------------------------------------------------------------
// CDNA5 async global->LDS path (ASYNCcnt-tracked), with safe fallback
// ---------------------------------------------------------------------------
#if defined(__has_builtin)
#  if __has_builtin(__builtin_amdgcn_global_load_async_to_lds_b128) && \
      __has_builtin(__builtin_amdgcn_s_wait_asynccnt)
#    define USE_ASYNC_LDS 1
#  endif
#endif
#ifndef USE_ASYNC_LDS
#  define USE_ASYNC_LDS 0
#endif

#if USE_ASYNC_LDS
typedef __attribute__((address_space(1))) v4i* gptr_v4i;   // global int4*
typedef __attribute__((address_space(3))) v4i* lptr_v4i;   // LDS    int4*
#endif

// copy 16 bytes (one b128) global -> LDS
__device__ __forceinline__ void cp16(const _Float16* g, _Float16* l) {
#if USE_ASYNC_LDS
  __builtin_amdgcn_global_load_async_to_lds_b128(
      (gptr_v4i)(v4i*)g, (lptr_v4i)(v4i*)l, 0, 0);
#else
  *(v8h*)l = *(const v8h*)g;
#endif
}

__device__ __forceinline__ void wait_async() {
#if USE_ASYNC_LDS
  __builtin_amdgcn_s_wait_asynccnt(0);
#endif
}

// ---------------------------------------------------------------------------
// Problem constants and workspace layout (in _Float16 units)
// ---------------------------------------------------------------------------
constexpr int    NPIX = 9216;               // 96*96 tokens
constexpr int    CDIM = 128;
constexpr size_t NC   = (size_t)NPIX * CDIM; // 1,179,648

// masked inputs (f16, [n][c] row-major)
constexpr size_t XM_OBJ = 0;
constexpr size_t FM_OBJ = 1 * NC;
constexpr size_t XM_BG  = 2 * NC;
constexpr size_t FM_BG  = 3 * NC;
// Q/K row-major [n][c]; V transposed [c][n] for contiguous PV B-fragments
constexpr size_t Q_OBJ  = 4 * NC;
constexpr size_t K_OBJ  = 5 * NC;
constexpr size_t VT_OBJ = 6 * NC;
constexpr size_t Q_BG   = 7 * NC;
constexpr size_t K_BG   = 8 * NC;
constexpr size_t VT_BG  = 9 * NC;
// aliases: masked inputs are dead after QKV projection
constexpr size_t O_OBJ  = XM_OBJ;  // attention output obj [n][c] f16
constexpr size_t O_BG   = FM_OBJ;  // attention output bg
constexpr size_t AM     = XM_BG;   // obj_att + bg_att (pre-conv) [n][c] f16
// f16 weights: 8 matrices [d][c]: qw_o,kw_o,vw_o,pw_o,qw_b,kw_b,vw_b,pw_b
constexpr size_t W16    = 10 * NC;
constexpr size_t WCONV  = W16 + 8 * 16384;   // 9 taps of [oc][ic]
// total = 10*NC + 17*16384 halves  ~= 24.2 MB

// LDS tile geometry (pads keep 16B alignment and stagger banks)
constexpr int KSTR = 136;                   // K tile row stride (128+8)
constexpr int VSTR = 40;                    // V tile row stride (32+8)
constexpr int KTILE = 32 * KSTR;            // halves per K buffer
constexpr int VTILE = 128 * VSTR;           // halves per V buffer

// ---------------------------------------------------------------------------
// Kernel 1: mask + transpose inputs to f16 [n][c]
// ---------------------------------------------------------------------------
__global__ void __launch_bounds__(256) prep_data_kernel(
    const float* __restrict__ x, const float* __restrict__ fm,
    const float* __restrict__ hm, _Float16* __restrict__ ws) {
  int idx = blockIdx.x * 256 + threadIdx.x;           // over CDIM*NPIX, c-major
  if (idx >= CDIM * NPIX) return;
  int c = idx / NPIX, n = idx % NPIX;
  float m  = hm[n];
  float mo = (m > 0.3f) ? 1.0f : 0.01f;
  float mb = (m > 0.3f) ? 0.01f : 1.0f;
  float xv = x[idx], fv = fm[idx];
  size_t o = (size_t)n * CDIM + c;
  ws[XM_OBJ + o] = (_Float16)(xv * mo);
  ws[XM_BG  + o] = (_Float16)(xv * mb);
  ws[FM_OBJ + o] = (_Float16)(fv * mo);
  ws[FM_BG  + o] = (_Float16)(fv * mb);
}

// ---------------------------------------------------------------------------
// Kernel 2: convert weights to f16; reorder conv weights to 9 [oc][ic] mats
// ---------------------------------------------------------------------------
__global__ void __launch_bounds__(256) prep_weights_kernel(
    const float* __restrict__ qo, const float* __restrict__ ko,
    const float* __restrict__ vo, const float* __restrict__ po,
    const float* __restrict__ qb, const float* __restrict__ kb,
    const float* __restrict__ vb, const float* __restrict__ pb,
    const float* __restrict__ convw, _Float16* __restrict__ ws) {
  int idx = blockIdx.x * 256 + threadIdx.x;
  if (idx < 8 * 16384) {
    int m = idx >> 14, e = idx & 16383;
    float v;
    switch (m) {
      case 0: v = qo[e]; break;  case 1: v = ko[e]; break;
      case 2: v = vo[e]; break;  case 3: v = po[e]; break;
      case 4: v = qb[e]; break;  case 5: v = kb[e]; break;
      case 6: v = vb[e]; break;  default: v = pb[e]; break;
    }
    ws[W16 + idx] = (_Float16)v;
  } else {
    int i2 = idx - 8 * 16384;                  // over 128*128*9
    if (i2 < 128 * 128 * 9) {
      int oc = i2 / (128 * 9), rem = i2 % (128 * 9);
      int ic = rem / 9, tap = rem % 9;
      ws[WCONV + (size_t)tap * 16384 + oc * 128 + ic] = (_Float16)convw[i2];
    }
  }
}

// ---------------------------------------------------------------------------
// Shared GEMM helper: one wave computes 16 rows x 128 cols, K=128.
// A row-major f16 [*,128]; W f16 [d][c] (used as B[k=c][col=d]).
// ---------------------------------------------------------------------------
__device__ __forceinline__ void gemm_tile_16x128(
    const _Float16* __restrict__ A, const _Float16* __restrict__ W,
    int row0, int lane, v8f c[8]) {
  const int r = lane & 15, g = lane >> 4;
  #pragma unroll
  for (int kc = 0; kc < 4; ++kc) {
    V16 a;
    const _Float16* ap = A + (size_t)(row0 + r) * 128 + kc * 32 + 16 * g;
    a.h[0] = *(const v8h*)ap;            // K = kc*32 + 16g + 0..7
    a.h[1] = *(const v8h*)(ap + 8);      // K = kc*32 + 16g + 8..15
    #pragma unroll
    for (int t = 0; t < 8; ++t) {
      v16h b = ld16(W + (size_t)(t * 16 + r) * 128 + kc * 32 + 16 * g);
      c[t] = wmma_f32(a.v, b, c[t]);
    }
  }
}

// ---------------------------------------------------------------------------
// Kernel 3: QKV projections (6 GEMMs selected by blockIdx.y). V stored ^T.
// ---------------------------------------------------------------------------
__global__ void __launch_bounds__(256) qkv_kernel(_Float16* __restrict__ ws) {
  const int tid = threadIdx.x, wave = tid >> 5, lane = tid & 31;
  const int r = lane & 15, g = lane >> 4;
  const int row0 = (blockIdx.x * 8 + wave) * 16;
  size_t aoff, doff; int wsel; bool tr;
  switch (blockIdx.y) {
    case 0:  aoff = XM_OBJ; wsel = 0; doff = Q_OBJ;  tr = false; break;
    case 1:  aoff = FM_OBJ; wsel = 1; doff = K_OBJ;  tr = false; break;
    case 2:  aoff = XM_OBJ; wsel = 2; doff = VT_OBJ; tr = true;  break;
    case 3:  aoff = XM_BG;  wsel = 4; doff = Q_BG;   tr = false; break;
    case 4:  aoff = FM_BG;  wsel = 5; doff = K_BG;   tr = false; break;
    default: aoff = XM_BG;  wsel = 6; doff = VT_BG;  tr = true;  break;
  }
  v8f c[8] = {};
  gemm_tile_16x128(ws + aoff, ws + W16 + (size_t)wsel * 16384, row0, lane, c);
  _Float16* D = ws + doff;
  #pragma unroll
  for (int t = 0; t < 8; ++t) {
    int d = t * 16 + r;
    #pragma unroll
    for (int j = 0; j < 8; ++j) {
      int n = row0 + j + 8 * g;
      if (tr) D[(size_t)d * NPIX + n] = (_Float16)c[t][j];
      else    D[(size_t)n * CDIM + d] = (_Float16)c[t][j];
    }
  }
}

// ---------------------------------------------------------------------------
// Kernel 4: flash attention. One wave = 16-row query tile; block = 8 waves.
// Key blocks of 32 staged in LDS, shared by all 8 waves (8x less L2 BW),
// DOUBLE-BUFFERED via CDNA5 global_load_async_to_lds_b128 (ASYNCcnt) so the
// next tile's global->LDS DMA overlaps this tile's WMMA work.
// Online softmax in f32; P transposed to A-layout via per-wave LDS scratch.
// ---------------------------------------------------------------------------
__global__ void __launch_bounds__(256) flash_attn_kernel(_Float16* __restrict__ ws) {
  __shared__ _Float16 klds[2 * KTILE];       // K tiles [2][32][128+8]
  __shared__ _Float16 vlds[2 * VTILE];       // V tiles [2][128][32+8]
  __shared__ _Float16 plds[8 * 16 * 40];     // per-wave P scratch [16][32+8]
  const int tid = threadIdx.x, wave = tid >> 5, lane = tid & 31;
  const int r = lane & 15, g = lane >> 4;
  const int q0 = (blockIdx.x * 8 + wave) * 16;
  const int br = blockIdx.y;
  const _Float16* Qp = ws + (br ? Q_BG  : Q_OBJ);
  const _Float16* Kp = ws + (br ? K_BG  : K_OBJ);
  const _Float16* Vt = ws + (br ? VT_BG : VT_OBJ);
  _Float16*       Op = ws + (br ? O_BG  : O_OBJ);

  // stage one 32-key K/V tile into LDS buffer `buf` (all 256 threads help)
  auto stage_tiles = [&](int buf, int key0) {
    {   // K tile: 32 rows x 256B, 8 threads/row x 32B (2 async b128 each)
      int k = tid >> 3, off = (tid & 7) * 16;
      const _Float16* src = Kp + (size_t)(key0 + k) * 128 + off;
      _Float16* dst = klds + buf * KTILE + k * KSTR + off;
      cp16(src, dst);
      cp16(src + 8, dst + 8);
    }
    {   // V^T tile: 128 rows x 64B, 2 threads/row x 32B
      int d = tid >> 1, off = (tid & 1) * 16;
      const _Float16* src = Vt + (size_t)d * NPIX + key0 + off;
      _Float16* dst = vlds + buf * VTILE + d * VSTR + off;
      cp16(src, dst);
      cp16(src + 8, dst + 8);
    }
  };

  // Q fragments for all 4 K-chunks, pre-scaled by SCALE = 0.25 (exact in f16)
  V16 qa[4];
  #pragma unroll
  for (int kc = 0; kc < 4; ++kc) {
    const _Float16* qp = Qp + (size_t)(q0 + r) * 128 + kc * 32 + 16 * g;
    qa[kc].h[0] = *(const v8h*)qp;
    qa[kc].h[1] = *(const v8h*)(qp + 8);
    #pragma unroll
    for (int e = 0; e < 16; ++e) qa[kc].v[e] = qa[kc].v[e] * (_Float16)0.25f;
  }

  v8f o[8] = {};
  float mrow[8], lrow[8];
  #pragma unroll
  for (int j = 0; j < 8; ++j) { mrow[j] = -1e30f; lrow[j] = 0.0f; }
  _Float16* P = plds + wave * (16 * 40);

  stage_tiles(0, 0);                         // prologue: tile 0 in flight

  for (int kb = 0; kb < NPIX / 32; ++kb) {
    const int cur = kb & 1;
    // Wait for this wave's outstanding async fills (tile kb), then barrier:
    // after it, tile kb is complete in LDS AND every wave has finished
    // reading buffer cur^1 (its compute for kb-1 precedes this barrier).
    wait_async();
    __syncthreads();
    if (kb + 1 < NPIX / 32) stage_tiles(cur ^ 1, (kb + 1) * 32);

    const _Float16* kl = klds + cur * KTILE;
    const _Float16* vl = vlds + cur * VTILE;

    // S = Q * K^T  (16 x 32, two 16x16 C-tiles), f32, scale folded into Q
    v8f s0 = {}, s1 = {};
    #pragma unroll
    for (int kc = 0; kc < 4; ++kc) {
      v16h b0 = ld16(kl + (size_t)r        * KSTR + kc * 32 + 16 * g);
      v16h b1 = ld16(kl + (size_t)(16 + r) * KSTR + kc * 32 + 16 * g);
      s0 = wmma_f32(qa[kc].v, b0, s0);
      s1 = wmma_f32(qa[kc].v, b1, s1);
    }

    // Online softmax over the 32 new columns; C-tile row = j + 8g.
    float cf[8];
    #pragma unroll
    for (int j = 0; j < 8; ++j) {
      float a0 = s0[j], a1 = s1[j];
      float mx = fmaxf(a0, a1);
      mx = fmaxf(mx, __shfl_xor(mx, 1, 32));
      mx = fmaxf(mx, __shfl_xor(mx, 2, 32));
      mx = fmaxf(mx, __shfl_xor(mx, 4, 32));
      mx = fmaxf(mx, __shfl_xor(mx, 8, 32));   // reduce within 16-lane half
      float mn  = fmaxf(mrow[j], mx);
      float cfj = __expf(mrow[j] - mn);
      float p0  = __expf(a0 - mn);
      float p1  = __expf(a1 - mn);
      float rs  = p0 + p1;
      rs += __shfl_xor(rs, 1, 32);
      rs += __shfl_xor(rs, 2, 32);
      rs += __shfl_xor(rs, 4, 32);
      rs += __shfl_xor(rs, 8, 32);
      lrow[j] = lrow[j] * cfj + rs;
      mrow[j] = mn;
      cf[j]   = cfj;
      int row = j + 8 * g;                     // transpose P to [row][col] f16
      P[row * 40 + r]      = (_Float16)p0;
      P[row * 40 + 16 + r] = (_Float16)p1;
    }
    #pragma unroll
    for (int t = 0; t < 8; ++t) {
      #pragma unroll
      for (int j = 0; j < 8; ++j) o[t][j] *= cf[j];
    }

    // Reload P in A-fragment layout (same wave: DScnt-ordered, no barrier)
    V16 pa;
    pa.h[0] = *(const v8h*)(P + r * 40 + 16 * g);
    pa.h[1] = *(const v8h*)(P + r * 40 + 8 + 16 * g);

    // O += P(16x32) * V(32x128): 8 WMMAs from the shared V tile
    #pragma unroll
    for (int t = 0; t < 8; ++t) {
      v16h bv = ld16(vl + (size_t)(t * 16 + r) * VSTR + 16 * g);
      o[t] = wmma_f32(pa.v, bv, o[t]);
    }
  }

  // normalize and store O as f16 [n][d]
  float inv[8];
  #pragma unroll
  for (int j = 0; j < 8; ++j) inv[j] = 1.0f / lrow[j];
  #pragma unroll
  for (int t = 0; t < 8; ++t) {
    int d = t * 16 + r;
    #pragma unroll
    for (int j = 0; j < 8; ++j) {
      int n = q0 + j + 8 * g;
      Op[(size_t)n * CDIM + d] = (_Float16)(o[t][j] * inv[j]);
    }
  }
}

// ---------------------------------------------------------------------------
// Kernel 5: output projection of both branches + bias, summed into AM (f16)
// ---------------------------------------------------------------------------
__global__ void __launch_bounds__(256) proj_out_kernel(
    _Float16* __restrict__ ws, const float* __restrict__ pbo,
    const float* __restrict__ pbb) {
  const int tid = threadIdx.x, wave = tid >> 5, lane = tid & 31;
  const int r = lane & 15, g = lane >> 4;
  const int row0 = (blockIdx.x * 8 + wave) * 16;
  v8f c[8] = {};
  gemm_tile_16x128(ws + O_OBJ, ws + W16 + 3 * (size_t)16384, row0, lane, c);
  gemm_tile_16x128(ws + O_BG,  ws + W16 + 7 * (size_t)16384, row0, lane, c);
  _Float16* am = ws + AM;
  #pragma unroll
  for (int t = 0; t < 8; ++t) {
    int d = t * 16 + r;
    float bias = pbo[d] + pbb[d];
    #pragma unroll
    for (int j = 0; j < 8; ++j) {
      int n = row0 + j + 8 * g;
      am[(size_t)n * CDIM + d] = (_Float16)(c[t][j] + bias);
    }
  }
}

// ---------------------------------------------------------------------------
// Kernel 6: 3x3 SAME conv as 9 shifted implicit GEMMs; f32 NCHW output
// ---------------------------------------------------------------------------
__global__ void __launch_bounds__(256) conv_kernel(
    const _Float16* __restrict__ ws, const float* __restrict__ convb,
    float* __restrict__ out) {
  const int tid = threadIdx.x, wave = tid >> 5, lane = tid & 31;
  const int r = lane & 15, g = lane >> 4;
  const int row0 = (blockIdx.x * 8 + wave) * 16;
  const int nrow = row0 + r;                 // A-row owned by this lane
  const int h = nrow / 96, w = nrow % 96;
  const _Float16* am = ws + AM;
  const _Float16* wk = ws + WCONV;
  v8f c[8] = {};
  #pragma unroll
  for (int tap = 0; tap < 9; ++tap) {
    const int dh = tap / 3 - 1, dw = tap % 3 - 1;
    const int hh = h + dh, ww = w + dw;
    const bool valid = (hh >= 0) & (hh < 96) & (ww >= 0) & (ww < 96);
    const _Float16* arow = am + (size_t)(hh * 96 + ww) * CDIM;
    #pragma unroll
    for (int kc = 0; kc < 4; ++kc) {
      V16 a;
      if (valid) {
        const _Float16* ap = arow + kc * 32 + 16 * g;
        a.h[0] = *(const v8h*)ap;
        a.h[1] = *(const v8h*)(ap + 8);
      } else {
        #pragma unroll
        for (int e = 0; e < 16; ++e) a.v[e] = (_Float16)0.0f;
      }
      #pragma unroll
      for (int t = 0; t < 8; ++t) {   // EXEC all-ones again here
        v16h b = ld16(wk + (size_t)tap * 16384 +
                      (size_t)(t * 16 + r) * 128 + kc * 32 + 16 * g);
        c[t] = wmma_f32(a.v, b, c[t]);
      }
    }
  }
  #pragma unroll
  for (int t = 0; t < 8; ++t) {
    int d = t * 16 + r;
    float bias = convb[d];
    #pragma unroll
    for (int j = 0; j < 8; ++j) {
      int n = row0 + j + 8 * g;
      out[(size_t)d * NPIX + n] = c[t][j] + bias;   // NCHW: [c][h*w]
    }
  }
}

// ---------------------------------------------------------------------------
// Host launcher
// ---------------------------------------------------------------------------
extern "C" void kernel_launch(void* const* d_in, const int* in_sizes, int n_in,
                              void* d_out, int out_size, void* d_ws, size_t ws_size,
                              hipStream_t stream) {
  (void)in_sizes; (void)n_in; (void)out_size; (void)ws_size;
  const float* x   = (const float*)d_in[0];
  const float* fm  = (const float*)d_in[1];
  const float* hm  = (const float*)d_in[2];
  const float* qwo = (const float*)d_in[3];
  const float* kwo = (const float*)d_in[4];
  const float* vwo = (const float*)d_in[5];
  const float* pwo = (const float*)d_in[6];
  const float* pbo = (const float*)d_in[7];
  const float* qwb = (const float*)d_in[8];
  const float* kwb = (const float*)d_in[9];
  const float* vwb = (const float*)d_in[10];
  const float* pwb = (const float*)d_in[11];
  const float* pbb = (const float*)d_in[12];
  const float* cw  = (const float*)d_in[13];
  const float* cb  = (const float*)d_in[14];
  _Float16* ws  = (_Float16*)d_ws;   // needs ~24.2 MB (<< typical ws)
  float*    out = (float*)d_out;

  prep_data_kernel   <<<(CDIM * NPIX) / 256, 256, 0, stream>>>(x, fm, hm, ws);
  prep_weights_kernel<<<(8 * 16384 + 128 * 128 * 9) / 256, 256, 0, stream>>>(
      qwo, kwo, vwo, pwo, qwb, kwb, vwb, pwb, cw, ws);
  qkv_kernel         <<<dim3(NPIX / 128, 6), 256, 0, stream>>>(ws);
  flash_attn_kernel  <<<dim3(NPIX / 128, 2), 256, 0, stream>>>(ws);
  proj_out_kernel    <<<NPIX / 128, 256, 0, stream>>>(ws, pbo, pbb);
  conv_kernel        <<<NPIX / 128, 256, 0, stream>>>(ws, cb, out);
}
